// PhysicsLossTransient_83245056131253
// MI455X (gfx1250) — matrix operand order repaired and checked by reading, not compiled, not run on hardware
//
#include <hip/hip_runtime.h>

typedef __attribute__((ext_vector_type(16))) __bf16 v16bf;
typedef __attribute__((ext_vector_type(8)))  float  v8f;

#define NN      169     // 13*13 nodes
#define NPAD    192     // padded K-dim: 6 * 32
#define MTILES  11      // ceil(169/16) row tiles (176 rows)
#define KTILES  6       // 192 / 32
#define TB      128     // batch columns per workgroup
#define TVSTR   200     // LDS row stride (bf16 units) per batch column
#define NFRAGEL (MTILES*KTILES*32*16)   // bf16 elements of A-fragment scratch

// ---------------------------------------------------------------------------
// Prep: convert K (169x169 f32) into bf16 WMMA A-fragments in d_ws, laid out
// per the CDNA5 16-bit A-matrix (16x32) VGPR mapping so the GEMM loop can load
// each lane's 16 bf16 elements as one contiguous 32B read. Also zero d_out.
// A layout (ISA 7.12.2): lane half h=lane>>4, m=lane&15; element e -> VGPR
// v=e>>1; k = (v<4 ? 2v : 16+2(v-4)) + (e&1) + 8h.
// ---------------------------------------------------------------------------
__global__ void phys_prep_kernel(const float* __restrict__ K,
                                 __bf16* __restrict__ afrag,
                                 float* __restrict__ out) {
    int tid = blockIdx.x * 256 + threadIdx.x;
    if (blockIdx.x == 0 && threadIdx.x == 0) out[0] = 0.0f;
    if (tid >= NFRAGEL) return;
    int e    = tid & 15;
    int lane = (tid >> 4) & 31;
    int frag = tid >> 9;
    int mt = frag / KTILES;
    int kt = frag - mt * KTILES;
    int half = lane >> 4;
    int m    = lane & 15;
    int v    = e >> 1;
    int kloc = (v < 4 ? 2 * v : 16 + 2 * (v - 4)) + (e & 1) + half * 8;
    int row  = mt * 16 + m;        // node (output row of K@Tv)
    int col  = kt * 32 + kloc;     // summed node (K-dim)
    float val = (row < NN && col < NN) ? K[row * NN + col] : 0.0f;
    afrag[tid] = (__bf16)val;      // index == (frag*32 + lane)*16 + e
}

// ---------------------------------------------------------------------------
// Main: per block = 128 batch cols. Stage T_prev as bf16 in LDS, build B
// fragments per wave (16 cols each), run K@Tv on v_wmma_f32_16x16x32_bf16,
// finish the elementwise residual straight out of the accumulators, reduce.
// ---------------------------------------------------------------------------
__global__ __launch_bounds__(256)
void phys_loss_kernel(const float* __restrict__ Tp,  const float* __restrict__ H,
                      const float* __restrict__ Ifc, const float* __restrict__ Te,
                      const float* __restrict__ Tv,  const float* __restrict__ dtp,
                      const float* __restrict__ ediag,
                      const __bf16* __restrict__ afragG,
                      float* __restrict__ out, int batch) {
    __shared__ __bf16 tvs[TB * TVSTR];
    __shared__ float  red[8];

    const int  tid  = threadIdx.x;
    const int  lane = tid & 31;
    const int  wave = tid >> 5;
    const long base = (long)blockIdx.x * TB * NN;

    // ---- Phase A: T_prev -> bf16 LDS, layout [col][node], fully coalesced ----
    for (int idx = tid; idx < TB * NN; idx += 256) {
        int b = idx / NN;
        int n = idx - b * NN;
        tvs[b * TVSTR + n] = (__bf16)Tv[base + idx];
    }
    // zero the padded node rows 169..191 (fragment reads touch node < 192)
    for (int idx = tid; idx < TB * (NPAD - NN); idx += 256) {
        int b = idx / (NPAD - NN);
        int n = NN + (idx - b * (NPAD - NN));
        tvs[b * TVSTR + n] = (__bf16)0.0f;
    }
    __syncthreads();

    // ---- Phase B: build six 32x16 B fragments for this wave's 16 columns ----
    // B layout: N=lane&15; VGPR v packs K=(v<4 ? 2v : 16+2(v-4)) + 8*half
    // and K+1 (matches the A-fragment k order).
    const int half = lane >> 4;
    const int cb   = wave * 16 + (lane & 15);   // batch col within block tile
    union BF { v16bf v; unsigned u[8]; };
    BF bf[KTILES];
#pragma unroll
    for (int kt = 0; kt < KTILES; ++kt) {
#pragma unroll
        for (int v = 0; v < 8; ++v) {
            int node = kt * 32 + (v < 4 ? 2 * v : 16 + 2 * (v - 4)) + half * 8;  // even index
            bf[kt].u[v] = *(const unsigned*)&tvs[cb * TVSTR + node];
        }
    }

    const float dtv     = dtp[0];
    const float inv_dt  = 1.0f / dtv;
    const float dxy     = 0.1f / 12.0f;                        // DX == DY
    const float inv_den = 1.0f / (2700.0f * 900.0f * 0.001f * dxy * dxy);
    const float boltz   = 5.67e-08f;

    float s = 0.0f;
    const long gb = base + (long)cb * NN;   // this lane's batch row base

    for (int mt = 0; mt < MTILES; ++mt) {
        v8f acc = {0.f, 0.f, 0.f, 0.f, 0.f, 0.f, 0.f, 0.f};
#pragma unroll
        for (int kt = 0; kt < KTILES; ++kt) {
            v16bf a = *(const v16bf*)(afragG + ((mt * KTILES + kt) * 32 + lane) * 16);
            acc = __builtin_amdgcn_wmma_f32_16x16x32_bf16(
                false, a, false, bf[kt].v, (short)0, acc, false, false);
        }
        // C/D layout: lane = col, VGPR r -> node row mt*16 + r + 8*half
        const int nbase = mt * 16 + half * 8;
#pragma unroll
        for (int r = 0; r < 8; ++r) {
            int n = nbase + r;
            if (n < NN) {
                long g  = gb + n;
                float tp = Tp[g], tv = Tv[g], h = H[g], fi = Ifc[g], te = Te[g];
                float ed  = ediag[n];
                float tv2 = tv * tv, te2 = te * te;
                float rad = boltz * ed * (tv2 * tv2 - te2 * te2);
                float rhs = (h + fi - acc[r] - rad) * inv_den;
                float res = (tp - tv) * inv_dt - rhs;
                if (fi != 0.0f) res = 0.0f;     // interface mask
                s += fabsf(res);
            }
        }
    }

    // ---- Reduce: wave shuffle -> 8 partials -> one atomic per block ----
#pragma unroll
    for (int off = 16; off; off >>= 1) s += __shfl_xor(s, off, 32);
    if (lane == 0) red[wave] = s;
    __syncthreads();
    if (tid == 0) {
        float t = 0.0f;
#pragma unroll
        for (int w = 0; w < 8; ++w) t += red[w];
        atomicAdd(out, t / (169.0f * (float)batch));
    }
}

extern "C" void kernel_launch(void* const* d_in, const int* in_sizes, int n_in,
                              void* d_out, int out_size, void* d_ws, size_t ws_size,
                              hipStream_t stream) {
    const float* Tp  = (const float*)d_in[0];   // T_pred     [B,13,13]
    const float* H   = (const float*)d_in[1];   // heaters    [B,13,13]
    const float* Ifc = (const float*)d_in[2];   // interfaces [B,13,13]
    const float* Te  = (const float*)d_in[3];   // Tenv       [B,13,13]
    const float* Tv  = (const float*)d_in[4];   // T_prev     [B,13,13]
    const float* dt  = (const float*)d_in[5];   // scalar
    const float* K   = (const float*)d_in[6];   // [169,169]
    const float* ed  = (const float*)d_in[7];   // [169]
    float*  out   = (float*)d_out;
    __bf16* afrag = (__bf16*)d_ws;              // 67,584 bytes used

    int batch = in_sizes[0] / NN;               // 131072

    phys_prep_kernel<<<(NFRAGEL + 255) / 256, 256, 0, stream>>>(K, afrag, out);
    phys_loss_kernel<<<batch / TB, 256, 0, stream>>>(Tp, H, Ifc, Te, Tv, dt, ed,
                                                     afrag, out, batch);
}
